// RNNTGreedyDecoder_18915035971758
// MI455X (gfx1250) — matrix operand order repaired and checked by reading, not compiled, not run on hardware
//
#include <hip/hip_runtime.h>
#include <hip/hip_bf16.h>
#include <stdint.h>

// ---------------------------------------------------------------------------
// RNN-T greedy decoder for MI455X (gfx1250, wave32, WMMA bf16)
// B=8, Fin=80, T=256, H=768, V=4096, BLANK=4095, MAX_SYM=2
// Round 3: defeat LICM weight-stream hoisting (round-2 disasm showed the
// loop-invariant weight loads hoisted to the preheader and spilled to
// scratch).  A runtime-zero offset (kernel arg -> LDS -> re-read every
// iteration) makes weight addresses loop-variant so B-fragments stream from
// L2 via global_load_b128 inside the loop.
// ---------------------------------------------------------------------------

typedef __attribute__((ext_vector_type(16))) __bf16 v16bf;
typedef __attribute__((ext_vector_type(8)))  float  v8f;

union FragU { unsigned int u[8]; uint4 q[2]; v16bf v; };

__device__ __forceinline__ unsigned int f2bf(float x) {
    unsigned int u = __float_as_uint(x);
    unsigned int r = ((u >> 16) & 1u) + 0x7FFFu;   // round-to-nearest-even
    return (u + r) >> 16;
}
__device__ __forceinline__ float sigf(float x)   { return 1.f / (1.f + __expf(-x)); }
__device__ __forceinline__ float tanhfa(float x) { return 1.f - 2.f / (__expf(2.f * x) + 1.f); }

// Fragment layout (A, B and packed weights): 256 dwords per 16x32 bf16 tile,
// per-lane contiguous: dword index = lane*8 + v  ->  two b128 loads per lane.
__device__ __forceinline__ v16bf frag_ld(const unsigned int* p, int lane) {
    FragU f;
    const uint4* q = (const uint4*)p;
    f.q[0] = q[lane * 2 + 0];
    f.q[1] = q[lane * 2 + 1];
    return f.v;
}

__device__ __forceinline__ v8f wmma_bf16(v16bf a, v16bf b, v8f c) {
    return __builtin_amdgcn_wmma_f32_16x16x32_bf16(false, a, false, b, (short)0, c,
                                                   false, false);
}

// ---------------------------------------------------------------------------
// Pack fp32 weight W[O,I] (used as out = x @ W^T) into WMMA B-fragment tiles.
// Fragment (nt, kt): dword (lane*8+v) holds bf16 pair (W[n,k], W[n,k+1]),
//   n = nt*16 + lane%16,   k = kt*32 + 2*(v%4) + 8*(lane/16) + 16*(v/4)
// ---------------------------------------------------------------------------
__global__ void pack_b_kernel(const float* __restrict__ W, unsigned int* __restrict__ dst,
                              int O, int I, int KT, int NT) {
    int tid = blockIdx.x * blockDim.x + threadIdx.x;
    int total = NT * KT * 256;
    if (tid >= total) return;
    int nt  = tid / (KT * 256);
    int rem = tid - nt * (KT * 256);
    int kt  = rem >> 8;
    int idx = rem & 255;
    int ln  = idx >> 3;
    int v   = idx & 7;
    int n = nt * 16 + (ln & 15);
    int k = kt * 32 + 2 * (v & 3) + 8 * (ln >> 4) + 16 * (v >> 2);
    unsigned lo = 0, hi = 0;
    if (n < O) {
        if (k     < I) lo = f2bf(W[(size_t)n * I + k]);
        if (k + 1 < I) hi = f2bf(W[(size_t)n * I + k + 1]);
    }
    dst[tid] = lo | (hi << 16);
}

// emb fp32 [4095,768] -> bf16 pairs, row-major dwords (dword j = cols 2j,2j+1)
__global__ void pack_emb_kernel(const float* __restrict__ emb, unsigned int* __restrict__ dst,
                                int total) {
    int tid = blockIdx.x * blockDim.x + threadIdx.x;
    if (tid >= total) return;
    int row = tid / 384;
    int c2  = tid - row * 384;
    unsigned lo = f2bf(emb[(size_t)row * 768 + 2 * c2]);
    unsigned hi = f2bf(emb[(size_t)row * 768 + 2 * c2 + 1]);
    dst[tid] = lo | (hi << 16);
}

// ---------------------------------------------------------------------------
// Encoder LSTM: one persistent workgroup, 16 waves.  Wave w owns hidden tiles
// {3w,3w+1,3w+2}; each processed with 4 gate accumulators and an innermost K
// loop (1 LDS A-frag + 4 global B-frags live).  c-state in VGPRs, h2 buffered
// in registers and scattered after the barrier.
// ---------------------------------------------------------------------------
__global__ __launch_bounds__(512, 1) void rnnt_encoder(
    const float* __restrict__ audio, const float* __restrict__ b_e,
    const unsigned int* __restrict__ wihe, const unsigned int* __restrict__ whhe,
    unsigned short* __restrict__ fs_out, int dz) {
    __shared__ __align__(16) unsigned int afrag[27 * 256];   // 3 x-frags + 24 h-frags
    __shared__ int dynz;

    const int tid  = threadIdx.x;
    const int lane = tid & 31;
    const int w    = tid >> 5;

    for (int i = tid; i < 27 * 256; i += 512) afrag[i] = 0;
    if (tid == 0) dynz = dz;   // runtime 0: re-read each iteration (anti-LICM)

    float be_r[3][4];
#pragma unroll
    for (int ht = 0; ht < 3; ++ht)
#pragma unroll
        for (int g = 0; g < 4; ++g)
            be_r[ht][g] = b_e[g * 768 + (w * 3 + ht) * 16 + (lane & 15)];

    float cst[3][8];
#pragma unroll
    for (int ht = 0; ht < 3; ++ht)
#pragma unroll
        for (int j = 0; j < 8; ++j) cst[ht][j] = 0.f;

    for (int t = 0; t < 256; ++t) {
        __syncthreads();
        // build x_t A-fragments (Fin=80 padded to 96 -> 3 K-tiles)
        if (tid < 768) {
            int kt = tid >> 8, idx = tid & 255;
            int ln = idx >> 3, v = idx & 7;
            int b = ln & 15;
            int k = kt * 32 + 2 * (v & 3) + 8 * (ln >> 4) + 16 * (v >> 2);
            unsigned lo = 0, hi = 0;
            if (b < 8) {
                if (k < 80)     lo = f2bf(audio[(b * 80 + k) * 256 + t]);
                if (k + 1 < 80) hi = f2bf(audio[(b * 80 + k + 1) * 256 + t]);
            }
            afrag[kt * 256 + idx] = lo | (hi << 16);
        }
        __syncthreads();

        // loop-variant (runtime-zero) base offset: keeps weight loads in-loop
        const int dyn = dynz;
        const unsigned int* wihe_d = wihe + dyn;
        const unsigned int* whhe_d = whhe + dyn;

        float h2buf[3][8];
#pragma unroll
        for (int ht = 0; ht < 3; ++ht) {
            const int htg = w * 3 + ht;
            v8f acc0 = {}, acc1 = {}, acc2 = {}, acc3 = {};
            // x @ W_ih_e^T   (3 K-tiles)
            for (int kf = 0; kf < 3; ++kf) {
                v16bf a = frag_ld(afrag + kf * 256, lane);
                const unsigned int* w0 = wihe_d + ((size_t)((0 * 48 + htg) * 3 + kf) << 8);
                const unsigned int* w1 = wihe_d + ((size_t)((1 * 48 + htg) * 3 + kf) << 8);
                const unsigned int* w2 = wihe_d + ((size_t)((2 * 48 + htg) * 3 + kf) << 8);
                const unsigned int* w3 = wihe_d + ((size_t)((3 * 48 + htg) * 3 + kf) << 8);
                acc0 = wmma_bf16(a, frag_ld(w0, lane), acc0);
                acc1 = wmma_bf16(a, frag_ld(w1, lane), acc1);
                acc2 = wmma_bf16(a, frag_ld(w2, lane), acc2);
                acc3 = wmma_bf16(a, frag_ld(w3, lane), acc3);
            }
            // h @ W_hh_e^T   (24 K-tiles)
            for (int kf = 0; kf < 24; ++kf) {
                v16bf a = frag_ld(afrag + (3 + kf) * 256, lane);
                const unsigned int* w0 = whhe_d + ((size_t)((0 * 48 + htg) * 24 + kf) << 8);
                const unsigned int* w1 = whhe_d + ((size_t)((1 * 48 + htg) * 24 + kf) << 8);
                const unsigned int* w2 = whhe_d + ((size_t)((2 * 48 + htg) * 24 + kf) << 8);
                const unsigned int* w3 = whhe_d + ((size_t)((3 * 48 + htg) * 24 + kf) << 8);
                __builtin_prefetch(w0 + 256, 0, 1);
                acc0 = wmma_bf16(a, frag_ld(w0, lane), acc0);
                acc1 = wmma_bf16(a, frag_ld(w1, lane), acc1);
                acc2 = wmma_bf16(a, frag_ld(w2, lane), acc2);
                acc3 = wmma_bf16(a, frag_ld(w3, lane), acc3);
            }
#pragma unroll
            for (int j = 0; j < 8; ++j) {
                float iv = acc0[j] + be_r[ht][0];
                float fv = acc1[j] + be_r[ht][1];
                float gv = acc2[j] + be_r[ht][2];
                float ov = acc3[j] + be_r[ht][3];
                float c2 = sigf(fv) * cst[ht][j] + sigf(iv) * tanhfa(gv);
                cst[ht][j]   = c2;
                h2buf[ht][j] = sigf(ov) * tanhfa(c2);
            }
        }
        __syncthreads();   // all waves done reading h-frags for this t

        // scatter h2 into LDS A-fragments (next step) and fs (global)
#pragma unroll
        for (int ht = 0; ht < 3; ++ht) {
            int htg  = w * 3 + ht;
            int col  = lane & 15;
            int krel = ((htg & 1) << 4) | col;
            int la16 = (krel >> 3) & 1;
            int va   = 4 * (krel >> 4) + ((krel & 7) >> 1);
            int hf   = krel & 1;
#pragma unroll
            for (int j = 0; j < 8; ++j) {
                int m   = j + 8 * (lane >> 4);
                int laf = m + 16 * la16;
                unsigned short hb = (unsigned short)f2bf(h2buf[ht][j]);
                ((unsigned short*)afrag)[((3 + (htg >> 1)) * 256 + laf * 8 + va) * 2 + hf] = hb;
                fs_out[((size_t)t * 6144 + (size_t)(htg >> 1) * 256 + laf * 8 + va) * 2 + hf] = hb;
            }
        }
    }
}

// ---------------------------------------------------------------------------
// Greedy decode: batched across B in one persistent workgroup.
// ---------------------------------------------------------------------------
__global__ __launch_bounds__(512, 1) void rnnt_decode(
    const int* __restrict__ lens, const float* __restrict__ b_p,
    const float* __restrict__ b_j,
    const unsigned int* __restrict__ wihp, const unsigned int* __restrict__ whhp,
    const unsigned int* __restrict__ wj, const unsigned int* __restrict__ embbf,
    const unsigned int* __restrict__ fs,
    int* __restrict__ labels, int* __restrict__ counts, float* __restrict__ scores,
    int dz) {
    __shared__ __align__(16) unsigned int afrag_h[24 * 256];   // pred h state
    __shared__ __align__(16) unsigned int afrag_xj[24 * 256];  // xe, then joint-h2
    __shared__ float st_m[16][16];
    __shared__ float st_s[16][16];
    __shared__ int   st_i[16][16];
    __shared__ int   last_s[16], cnt_s[16], emitF[16], activeF[16], lenS[16];
    __shared__ int   fin_idx[16];
    __shared__ float fin_sc[16];
    __shared__ int   dynz;

    const int tid  = threadIdx.x;
    const int lane = tid & 31;
    const int w    = tid >> 5;

    for (int i = tid; i < 24 * 256; i += 512) { afrag_h[i] = 0; afrag_xj[i] = 0; }
    if (tid < 16) {
        last_s[tid] = -1; cnt_s[tid] = 0; emitF[tid] = 0; activeF[tid] = 0;
        lenS[tid] = (tid < 8) ? lens[tid] : 0;
    }
    if (tid == 0) dynz = dz;   // runtime 0: re-read each iteration (anti-LICM)
    for (int i = tid; i < 8 * 512; i += 512) labels[i] = 4095;   // blank-pad

    float bp_r[3][4];
#pragma unroll
    for (int ht = 0; ht < 3; ++ht)
#pragma unroll
        for (int g = 0; g < 4; ++g)
            bp_r[ht][g] = b_p[g * 768 + (w * 3 + ht) * 16 + (lane & 15)];

    float cst[3][8];
#pragma unroll
    for (int ht = 0; ht < 3; ++ht)
#pragma unroll
        for (int j = 0; j < 8; ++j) cst[ht][j] = 0.f;

    __syncthreads();

    for (int t = 0; t < 256; ++t) {
        for (int s = 0; s < 2; ++s) {
            // (a) per-row activity + build xe A-fragments from embedding
            if (s == 0 && tid < 8) activeF[tid] = (t < lenS[tid]) ? 1 : 0;
            for (int i = tid; i < 24 * 256; i += 512) {
                int kf = i >> 8, idx = i & 255;
                int ln = idx >> 3, v = idx & 7;
                int b = ln & 15;
                unsigned dv = 0;
                int ls = last_s[b];
                if (b < 8 && ls >= 0) {
                    int k = kf * 32 + 2 * (v & 3) + 8 * (ln >> 4) + 16 * (v >> 2);
                    dv = embbf[(size_t)ls * 384 + (k >> 1)];
                }
                afrag_xj[i] = dv;
            }
            __syncthreads();

            // loop-variant (runtime-zero) base offset for all weight streams
            const int dyn = dynz;
            const unsigned int* wihp_d = wihp + dyn;
            const unsigned int* whhp_d = whhp + dyn;
            const unsigned int* wj_d   = wj + dyn;

            // (c) pred-LSTM gates: one hidden tile (4 gate accs) at a time
            float c2a[3][8], h2a[3][8];
#pragma unroll
            for (int ht = 0; ht < 3; ++ht) {
                const int htg = w * 3 + ht;
                v8f acc0 = {}, acc1 = {}, acc2 = {}, acc3 = {};
                for (int kf = 0; kf < 24; ++kf) {         // xe @ W_ih_p^T
                    v16bf a = frag_ld(afrag_xj + kf * 256, lane);
                    const unsigned int* w0 = wihp_d + ((size_t)((0 * 48 + htg) * 24 + kf) << 8);
                    const unsigned int* w1 = wihp_d + ((size_t)((1 * 48 + htg) * 24 + kf) << 8);
                    const unsigned int* w2 = wihp_d + ((size_t)((2 * 48 + htg) * 24 + kf) << 8);
                    const unsigned int* w3 = wihp_d + ((size_t)((3 * 48 + htg) * 24 + kf) << 8);
                    __builtin_prefetch(w0 + 256, 0, 1);
                    acc0 = wmma_bf16(a, frag_ld(w0, lane), acc0);
                    acc1 = wmma_bf16(a, frag_ld(w1, lane), acc1);
                    acc2 = wmma_bf16(a, frag_ld(w2, lane), acc2);
                    acc3 = wmma_bf16(a, frag_ld(w3, lane), acc3);
                }
                for (int kf = 0; kf < 24; ++kf) {         // h @ W_hh_p^T
                    v16bf a = frag_ld(afrag_h + kf * 256, lane);
                    const unsigned int* w0 = whhp_d + ((size_t)((0 * 48 + htg) * 24 + kf) << 8);
                    const unsigned int* w1 = whhp_d + ((size_t)((1 * 48 + htg) * 24 + kf) << 8);
                    const unsigned int* w2 = whhp_d + ((size_t)((2 * 48 + htg) * 24 + kf) << 8);
                    const unsigned int* w3 = whhp_d + ((size_t)((3 * 48 + htg) * 24 + kf) << 8);
                    acc0 = wmma_bf16(a, frag_ld(w0, lane), acc0);
                    acc1 = wmma_bf16(a, frag_ld(w1, lane), acc1);
                    acc2 = wmma_bf16(a, frag_ld(w2, lane), acc2);
                    acc3 = wmma_bf16(a, frag_ld(w3, lane), acc3);
                }
#pragma unroll
                for (int j = 0; j < 8; ++j) {
                    float iv = acc0[j] + bp_r[ht][0];
                    float fv = acc1[j] + bp_r[ht][1];
                    float gv = acc2[j] + bp_r[ht][2];
                    float ov = acc3[j] + bp_r[ht][3];
                    float c2 = sigf(fv) * cst[ht][j] + sigf(iv) * tanhfa(gv);
                    c2a[ht][j] = c2;
                    h2a[ht][j] = sigf(ov) * tanhfa(c2);
                }
            }
            __syncthreads();

            // (e) scatter h2 into joint A-fragments (reuse xe buffer)
#pragma unroll
            for (int ht = 0; ht < 3; ++ht) {
                int htg  = w * 3 + ht;
                int col  = lane & 15;
                int krel = ((htg & 1) << 4) | col;
                int la16 = (krel >> 3) & 1;
                int va   = 4 * (krel >> 4) + ((krel & 7) >> 1);
                int hf   = krel & 1;
#pragma unroll
                for (int j = 0; j < 8; ++j) {
                    int m = j + 8 * (lane >> 4);
                    ((unsigned short*)afrag_xj)[((htg >> 1) * 256 + (m + 16 * la16) * 8 + va) * 2 + hf] =
                        (unsigned short)f2bf(h2a[ht][j]);
                }
            }
            __syncthreads();

            // (g) joint GEMM + online (max, argmax, sumexp) per row
            float run_m[8], run_s[8];
            int   run_i[8];
#pragma unroll
            for (int j = 0; j < 8; ++j) { run_m[j] = -3.0e38f; run_s[j] = 0.f; run_i[j] = 0; }
            for (int nti = 0; nti < 16; ++nti) {
                int nt = w * 16 + nti;
                v8f acj = {};
                for (int kf = 0; kf < 24; ++kf) {
                    const unsigned int* wb = wj_d + ((size_t)(nt * 48 + kf) << 8);
                    __builtin_prefetch(wb + 256, 0, 1);
                    acj = wmma_bf16(frag_ld(fs + (size_t)t * 6144 + kf * 256, lane),
                                    frag_ld(wb, lane), acj);
                }
                for (int kf = 0; kf < 24; ++kf) {
                    const unsigned int* wb = wj_d + ((size_t)(nt * 48 + 24 + kf) << 8);
                    acj = wmma_bf16(frag_ld(afrag_xj + kf * 256, lane),
                                    frag_ld(wb, lane), acj);
                }
                int col = nt * 16 + (lane & 15);
                float bjv = b_j[col];
#pragma unroll
                for (int j = 0; j < 8; ++j) {
                    float val = acj[j] + bjv;
                    if (val > run_m[j]) {
                        run_s[j] = run_s[j] * __expf(run_m[j] - val) + 1.f;
                        run_m[j] = val; run_i[j] = col;
                    } else {
                        run_s[j] += __expf(val - run_m[j]);
                    }
                }
            }
            // butterfly across the 16 lanes of each half (rows stay in-half)
#pragma unroll
            for (int d = 1; d < 16; d <<= 1) {
#pragma unroll
                for (int j = 0; j < 8; ++j) {
                    float om = __shfl_xor(run_m[j], d, 32);
                    float os = __shfl_xor(run_s[j], d, 32);
                    int   oi = __shfl_xor(run_i[j], d, 32);
                    if (om > run_m[j]) {
                        run_s[j] = os + run_s[j] * __expf(run_m[j] - om);
                        run_m[j] = om; run_i[j] = oi;
                    } else {
                        run_s[j] += os * __expf(om - run_m[j]);
                    }
                }
            }
            if ((lane & 15) == 0) {
#pragma unroll
                for (int j = 0; j < 8; ++j) {
                    int row = j + 8 * (lane >> 4);
                    st_m[w][row] = run_m[j];
                    st_s[w][row] = run_s[j];
                    st_i[w][row] = run_i[j];
                }
            }
            __syncthreads();

            // (i) wave0: cross-wave merge, emit logic, outputs
            if (tid < 16) {
                float M = -3.0e38f, S = 0.f; int I2 = 0;
                for (int w2 = 0; w2 < 16; ++w2) {
                    float m2 = st_m[w2][tid], s2 = st_s[w2][tid];
                    int   i2 = st_i[w2][tid];
                    if (m2 > M) { S = s2 + S * __expf(M - m2); M = m2; I2 = i2; }
                    else        { S += s2 * __expf(m2 - M); }
                }
                fin_idx[tid] = I2;
                fin_sc[tid]  = -__logf(S);   // logp[argmax] = max - logsumexp
            }
            if (tid < 8) {
                int b   = tid;
                int idx = fin_idx[b];
                int act = activeF[b];
                scores[(b * 256 + t) * 2 + s] = act ? fin_sc[b] : 0.f;
                int em = (act && idx != 4095) ? 1 : 0;
                if (em) {
                    labels[b * 512 + cnt_s[b]] = idx;
                    cnt_s[b] += 1;
                    last_s[b] = idx;
                }
                emitF[b]   = em;
                activeF[b] = em;   // next sym step: active = emit & (t < len)
            }
            __syncthreads();

            // (k) masked state commit (rows 8..15 never emit -> stay zero)
#pragma unroll
            for (int ht = 0; ht < 3; ++ht) {
                int htg  = w * 3 + ht;
                int col  = lane & 15;
                int krel = ((htg & 1) << 4) | col;
                int la16 = (krel >> 3) & 1;
                int va   = 4 * (krel >> 4) + ((krel & 7) >> 1);
                int hf   = krel & 1;
#pragma unroll
                for (int j = 0; j < 8; ++j) {
                    int m = j + 8 * (lane >> 4);
                    if (emitF[m]) {
                        cst[ht][j] = c2a[ht][j];
                        ((unsigned short*)afrag_h)[((htg >> 1) * 256 + (m + 16 * la16) * 8 + va) * 2 + hf] =
                            (unsigned short)f2bf(h2a[ht][j]);
                    }
                }
            }
        }
    }
    __syncthreads();
    if (tid < 8) counts[tid] = cnt_s[tid];
}

// ---------------------------------------------------------------------------
extern "C" void kernel_launch(void* const* d_in, const int* in_sizes, int n_in,
                              void* d_out, int out_size, void* d_ws, size_t ws_size,
                              hipStream_t stream) {
    const float* audio = (const float*)d_in[0];
    const int*   lens  = (const int*)d_in[1];
    const float* emb   = (const float*)d_in[2];
    const float* Wihe  = (const float*)d_in[3];
    const float* Whhe  = (const float*)d_in[4];
    const float* b_e   = (const float*)d_in[5];
    const float* Wihp  = (const float*)d_in[6];
    const float* Whhp  = (const float*)d_in[7];
    const float* b_p   = (const float*)d_in[8];
    const float* Wj    = (const float*)d_in[9];
    const float* b_j   = (const float*)d_in[10];

    unsigned int* ws = (unsigned int*)d_ws;
    // workspace layout (dwords)
    const size_t oWihe = 0;                        // 192*3*256   = 147456
    const size_t oWhhe = oWihe + 147456;           // 192*24*256  = 1179648
    const size_t oWihp = oWhhe + 1179648;
    const size_t oWhhp = oWihp + 1179648;
    const size_t oWj   = oWhhp + 1179648;          // 256*48*256  = 3145728
    const size_t oEmb  = oWj + 3145728;            // 4095*384    = 1572480
    const size_t oFs   = oEmb + 1572480;           // 256*24*256  = 1572864
    const size_t total = oFs + 1572864;            // = 9977472 dwords (~40 MB)
    if (ws_size < total * 4) return;

    // 1) pack weights to bf16 WMMA B-fragments (streamed once; live in L2)
    pack_b_kernel<<<147456 / 256, 256, 0, stream>>>(Wihe, ws + oWihe, 3072, 80, 3, 192);
    pack_b_kernel<<<1179648 / 256, 256, 0, stream>>>(Whhe, ws + oWhhe, 3072, 768, 24, 192);
    pack_b_kernel<<<1179648 / 256, 256, 0, stream>>>(Wihp, ws + oWihp, 3072, 768, 24, 192);
    pack_b_kernel<<<1179648 / 256, 256, 0, stream>>>(Whhp, ws + oWhhp, 3072, 768, 24, 192);
    pack_b_kernel<<<3145728 / 256, 256, 0, stream>>>(Wj,   ws + oWj,   4096, 1536, 48, 256);
    pack_emb_kernel<<<(1572480 + 255) / 256, 256, 0, stream>>>(emb, ws + oEmb, 1572480);

    // 2) sequential encoder LSTM (persistent single workgroup, WMMA bf16)
    rnnt_encoder<<<1, 512, 0, stream>>>(audio, b_e, ws + oWihe, ws + oWhhe,
                                        (unsigned short*)(ws + oFs), 0);

    // 3) batched greedy decode (persistent single workgroup, WMMA bf16)
    int*   labels  = (int*)d_out;           // [8,512] blank-padded
    int*   counts  = (int*)d_out + 4096;    // [8]
    float* scoresp = (float*)d_out + 4104;  // [8,256,2]
    rnnt_decode<<<1, 512, 0, stream>>>(lens, b_p, b_j, ws + oWihp, ws + oWhhp,
                                       ws + oWj, ws + oEmb, ws + oFs,
                                       labels, counts, scoresp, 0);
}
